// GeometricAwareChannelRouter_27298812133589
// MI455X (gfx1250) — matrix operand order repaired and checked by reading, not compiled, not run on hardware
//
#include <hip/hip_runtime.h>

typedef unsigned short u16;
typedef __attribute__((ext_vector_type(16))) __bf16 v16bf;
typedef __attribute__((ext_vector_type(8)))  float  v8f;

// A/B fragment: 16 bf16 per lane = 2 x uint4 (two 16B LDS loads)
union Frag {
  v16bf v;
  uint4 q[2];
};

__device__ inline u16 f2bf(float f) {
  unsigned int u = __float_as_uint(f);
  u += 0x7FFFu + ((u >> 16) & 1u);   // round-to-nearest-even
  return (u16)(u >> 16);
}

__device__ inline v8f bf16_wmma(const Frag& a, const Frag& b, v8f c) {
  // D = A(16x32 bf16) * B(32x16 bf16) + C(16x16 f32)
  return __builtin_amdgcn_wmma_f32_16x16x32_bf16(
      false, a.v, false, b.v, (short)0, c, false, false);
}

// --- CDNA5 async global->LDS (ASYNCcnt-tracked), bypasses VGPRs ------------
__device__ inline void async_b128(void* lds_ptr, const void* gbase, unsigned byte_off) {
  unsigned lds = (unsigned)(size_t)lds_ptr;   // addrspace(3) byte offset
  asm volatile("global_load_async_to_lds_b128 %0, %1, %2 offset:0"
               :: "v"(lds), "v"(byte_off), "s"(gbase)
               : "memory");
}
__device__ inline void wait_async() {
  asm volatile("s_wait_asynccnt 0" ::: "memory");
}

// --- DPP16 butterfly reductions within each 16-lane row group (pure VALU) --
#define DPP_QUAD_XOR1 0xB1   // quad_perm [1,0,3,2]
#define DPP_QUAD_XOR2 0x4E   // quad_perm [2,3,0,1]
#define DPP_HALF_MIRR 0x141  // row_half_mirror
#define DPP_ROW_MIRR  0x140  // row_mirror
template <int CTRL>
__device__ inline float dpp_mov(float x) {
  int i = __float_as_int(x);
  int r = __builtin_amdgcn_update_dpp(i, i, CTRL, 0xF, 0xF, false);
  return __int_as_float(r);
}
__device__ inline float rowmax16(float x) {
  x = fmaxf(x, dpp_mov<DPP_QUAD_XOR1>(x));
  x = fmaxf(x, dpp_mov<DPP_QUAD_XOR2>(x));
  x = fmaxf(x, dpp_mov<DPP_HALF_MIRR>(x));
  x = fmaxf(x, dpp_mov<DPP_ROW_MIRR>(x));
  return x;
}
__device__ inline float rowsum16(float x) {
  x += dpp_mov<DPP_QUAD_XOR1>(x);
  x += dpp_mov<DPP_QUAD_XOR2>(x);
  x += dpp_mov<DPP_HALF_MIRR>(x);
  x += dpp_mov<DPP_ROW_MIRR>(x);
  return x;
}

// ---------------------------------------------------------------------------
// 1) Gate: pooled mean -> MLP(512->128->512) -> mask = (logit > 0)
// ---------------------------------------------------------------------------
__global__ __launch_bounds__(256) void gate_kernel(
    const float* __restrict__ x, const float* __restrict__ w1,
    const float* __restrict__ b1, const float* __restrict__ w2,
    const float* __restrict__ b2, float* __restrict__ mask) {
  const int b = blockIdx.x, t = threadIdx.x, lane = t & 31, wv = t >> 5;
  __shared__ float pooled[512];
  __shared__ float hid[128];

  for (int c = wv; c < 512; c += 8) {
    const float* row = x + ((size_t)(b * 512 + c)) * 1024;
    float s = 0.f;
    for (int i = lane; i < 1024; i += 32) s += row[i];
    s += __shfl_xor(s, 1);  s += __shfl_xor(s, 2);  s += __shfl_xor(s, 4);
    s += __shfl_xor(s, 8);  s += __shfl_xor(s, 16);
    if (lane == 0) pooled[c] = s * (1.0f / 1024.0f);
  }
  __syncthreads();
  if (t < 128) {
    float s = b1[t];
    for (int c = 0; c < 512; ++c) s += pooled[c] * w1[c * 128 + t];
    hid[t] = fmaxf(s, 0.f);
  }
  __syncthreads();
  for (int c = t; c < 512; c += 256) {
    float s = b2[c];
    for (int h = 0; h < 128; ++h) s += hid[h] * w2[h * 512 + c];
    mask[b * 512 + c] = (s > 0.f) ? 1.f : 0.f;   // sigmoid(s)>0.5 <=> s>0
  }
}

// ---------------------------------------------------------------------------
// 2) Prep: mask split + expmap0 (norm over W axis) + transpose to (B,N,C) bf16
// ---------------------------------------------------------------------------
__global__ __launch_bounds__(256) void prep_kernel(
    const float* __restrict__ x, const float* __restrict__ mask,
    u16* __restrict__ Ah, u16* __restrict__ Ae) {
  const int c0 = blockIdx.x * 32, h = blockIdx.y, b = blockIdx.z;
  const int t = threadIdx.x, lane = t & 31, wv = t >> 5;
  __shared__ u16 Th[32][33];
  __shared__ u16 Te[32][33];

#pragma unroll
  for (int i = 0; i < 4; ++i) {
    int cr = wv * 4 + i;
    float v = x[(((size_t)(b * 512 + c0 + cr)) * 32 + h) * 32 + lane];
    float ss = v * v;
    ss += __shfl_xor(ss, 1);  ss += __shfl_xor(ss, 2);  ss += __shfl_xor(ss, 4);
    ss += __shfl_xor(ss, 8);  ss += __shfl_xor(ss, 16);
    float n = sqrtf(ss);
    float m = mask[b * 512 + c0 + cr];
    float hs = m * tanhf(n) / fmaxf(n, 1e-15f);
    Th[cr][lane] = f2bf(v * hs);
    Te[cr][lane] = f2bf(v * (1.0f - m));
  }
  __syncthreads();
  {
    int c = t & 31, wg = t >> 5;
#pragma unroll
    for (int i = 0; i < 4; ++i) {
      int w = wg * 4 + i;
      size_t o = ((size_t)b * 1024 + h * 32 + w) * 512 + c0 + c;
      Ah[o] = Th[c][w];
      Ae[o] = Te[c][w];
    }
  }
}

// ---------------------------------------------------------------------------
// 3) GEMM: Y(8192 x JT) = A(8192 x 512, bf16) * Wt^T + bias
//    Wt is PRE-TRANSPOSED (JT x 512); tiles stream via async global->LDS.
//    128x128 block tile, 8 waves each 64x32 (4x2 wmma frags), K step 32.
// ---------------------------------------------------------------------------
template <int JT, bool OUTF32>
__global__ __launch_bounds__(256) void gemm_bias_kernel(
    const u16* __restrict__ A, const u16* __restrict__ Wt,
    const float* __restrict__ bias, void* __restrict__ Y) {
  const int t = threadIdx.x, lane = t & 31, wv = t >> 5;
  const int j0 = blockIdx.x * 128;
  const int m0 = blockIdx.y * 128;
  const int mb = (wv >> 2) * 64, jb = (wv & 3) * 32;
  const int m16 = lane & 15, kh = lane >> 4;

  __shared__ __align__(16) u16 As[128 * 40];   // [row][k] pad 8
  __shared__ __align__(16) u16 Ws[128 * 40];   // [j][k]   pad 8

  v8f acc[4][2];
#pragma unroll
  for (int mi = 0; mi < 4; ++mi)
#pragma unroll
    for (int ni = 0; ni < 2; ++ni) acc[mi][ni] = {};

  for (int k0 = 0; k0 < 512; k0 += 32) {
    __syncthreads();
    {
      int row = t >> 1, ch = t & 1;
      u16* dstA = &As[row * 40 + ch * 16];
      unsigned offA = (unsigned)((( (m0 + row) * 512 ) + k0 + ch * 16) * 2);
      async_b128(dstA,     A, offA);
      async_b128(dstA + 8, A, offA + 16);
      u16* dstW = &Ws[row * 40 + ch * 16];
      unsigned offW = (unsigned)((( (j0 + row) * 512 ) + k0 + ch * 16) * 2);
      async_b128(dstW,     Wt, offW);
      async_b128(dstW + 8, Wt, offW + 16);
    }
    wait_async();
    __syncthreads();

    Frag fa[4], fb[2];
#pragma unroll
    for (int mi = 0; mi < 4; ++mi) {  // A frag: K = kh*8+e / 16+kh*8+e
      const u16* rp = &As[(mb + mi * 16 + m16) * 40];
      fa[mi].q[0] = *(const uint4*)(rp + kh * 8);
      fa[mi].q[1] = *(const uint4*)(rp + 16 + kh * 8);
    }
#pragma unroll
    for (int ni = 0; ni < 2; ++ni) {  // B frag: lane n=m16, K = kh*8+e / 16+kh*8+e
      const u16* rp = &Ws[(jb + ni * 16 + m16) * 40];
      fb[ni].q[0] = *(const uint4*)(rp + kh * 8);
      fb[ni].q[1] = *(const uint4*)(rp + 16 + kh * 8);
    }
#pragma unroll
    for (int mi = 0; mi < 4; ++mi)
#pragma unroll
      for (int ni = 0; ni < 2; ++ni)
        acc[mi][ni] = bf16_wmma(fa[mi], fb[ni], acc[mi][ni]);
  }

#pragma unroll
  for (int mi = 0; mi < 4; ++mi)
#pragma unroll
    for (int ni = 0; ni < 2; ++ni) {
      int gj = j0 + jb + ni * 16 + m16;
      float bj = bias[gj];
#pragma unroll
      for (int r = 0; r < 8; ++r) {  // C layout: row = r + 8*kh, col = m16
        int gm = m0 + mb + mi * 16 + 8 * kh + r;
        float v = acc[mi][ni][r] + bj;
        if constexpr (OUTF32)
          ((float*)Y)[(size_t)gm * JT + gj] = v;
        else
          ((u16*)Y)[(size_t)gm * JT + gj] = f2bf(v);
      }
    }
}

// ---------------------------------------------------------------------------
// 4) Flash attention: per (b, head, 128-q-tile), 64-key tiles, online softmax.
//    QKV (8192 x 1536) bf16; Q already pre-scaled by 1/sqrt(64) via weights.
// ---------------------------------------------------------------------------
__global__ __launch_bounds__(256) void flash_kernel(
    const u16* __restrict__ QKV, u16* __restrict__ O) {
  const int t = threadIdx.x, lane = t & 31, wv = t >> 5;
  const int qt = blockIdx.x, head = blockIdx.y, b = blockIdx.z;
  const int m16 = lane & 15, kh = lane >> 4;
  const size_t rowQ = (size_t)b * 1024 + qt * 128;

  __shared__ __align__(16) u16 Qs[128 * 72];     // 128 q-rows x 64 d (pad 8)
  __shared__ __align__(16) u16 Ks[64 * 72];      // 64 keys  x 64 d (pad 8)
  __shared__ __align__(16) u16 Vt[64 * 72];      // V^T: 64 d x 64 keys (pad 8)
  __shared__ __align__(16) u16 Ps[8][16 * 64];   // per-wave P tile (16 x 64)

  {  // load Q tile via async global->LDS
    int row = t >> 1, ch = t & 1;
    unsigned base = (unsigned)(((rowQ + row) * 1536 + head * 64 + ch * 32) * 2);
    u16* dst = &Qs[row * 72 + ch * 32];
#pragma unroll
    for (int i = 0; i < 2; ++i) {
      async_b128(dst + i * 16,     QKV, base + i * 32);
      async_b128(dst + i * 16 + 8, QKV, base + i * 32 + 16);
    }
  }
  wait_async();
  __syncthreads();

  const int mb = wv * 16;
  Frag qf[2];  // Q rows in registers: d = 64 -> two K=32 frags
#pragma unroll
  for (int ks = 0; ks < 2; ++ks) {
    const u16* rp = &Qs[(mb + m16) * 72 + ks * 32];
    qf[ks].q[0] = *(const uint4*)(rp + kh * 8);
    qf[ks].q[1] = *(const uint4*)(rp + 16 + kh * 8);
  }

  float rm[8], rs[8];
  v8f o[4];
#pragma unroll
  for (int r = 0; r < 8; ++r) { rm[r] = -1e30f; rs[r] = 0.f; }
#pragma unroll
  for (int nt = 0; nt < 4; ++nt) o[nt] = {};

  for (int kt = 0; kt < 16; ++kt) {
    __syncthreads();
    if (t < 128) {  // K tile 64x64 via async (row-major == B^T for QK^T)
      int row = t >> 1, ch = t & 1;
      unsigned base =
          (unsigned)((((size_t)b * 1024 + kt * 64 + row) * 1536 + 512 + head * 64 + ch * 32) * 2);
      u16* dst = &Ks[row * 72 + ch * 32];
#pragma unroll
      for (int i = 0; i < 2; ++i) {
        async_b128(dst + i * 16,     QKV, base + i * 32);
        async_b128(dst + i * 16 + 8, QKV, base + i * 32 + 16);
      }
    } else {  // V tile transposed into Vt[d][key] (VGPR path: needs transpose)
      int tt = t - 128;
      int key = tt & 63, half = tt >> 6;
#pragma unroll
      for (int dcp = 0; dcp < 2; ++dcp) {
        int dc = half * 2 + dcp;
        const u16* src =
            QKV + ((size_t)b * 1024 + kt * 64 + key) * 1536 + 1024 + head * 64 + dc * 16;
        u16 tmp[16];
        *(uint4*)tmp = *(const uint4*)src;
        *(uint4*)(tmp + 8) = *(const uint4*)(src + 8);
#pragma unroll
        for (int e = 0; e < 16; ++e) Vt[(dc * 16 + e) * 72 + key] = tmp[e];
      }
    }
    wait_async();
    __syncthreads();

    // S = Q * K^T : four 16x16 j-tiles, two K=32 steps
    v8f s[4];
#pragma unroll
    for (int jt = 0; jt < 4; ++jt) s[jt] = {};
#pragma unroll
    for (int ks = 0; ks < 2; ++ks) {
#pragma unroll
      for (int jt = 0; jt < 4; ++jt) {
        Frag kf;
        const u16* rp = &Ks[(jt * 16 + m16) * 72 + ks * 32];
        kf.q[0] = *(const uint4*)(rp + kh * 8);
        kf.q[1] = *(const uint4*)(rp + 16 + kh * 8);
        s[jt] = bf16_wmma(qf[ks], kf, s[jt]);
      }
    }

    // online softmax: register r holds row (8*kh + r); cols across 16 lanes
#pragma unroll
    for (int r = 0; r < 8; ++r) {
      float mx = fmaxf(fmaxf(s[0][r], s[1][r]), fmaxf(s[2][r], s[3][r]));
      mx = rowmax16(mx);
      float mnew = fmaxf(rm[r], mx);
      float corr = __expf(rm[r] - mnew);
      float p0 = __expf(s[0][r] - mnew);
      float p1 = __expf(s[1][r] - mnew);
      float p2 = __expf(s[2][r] - mnew);
      float p3 = __expf(s[3][r] - mnew);
      float l = rowsum16((p0 + p1) + (p2 + p3));
      rs[r] = rs[r] * corr + l;
      rm[r] = mnew;
#pragma unroll
      for (int nt = 0; nt < 4; ++nt) o[nt][r] *= corr;
      u16* pr = &Ps[wv][(8 * kh + r) * 64 + m16];
      pr[0]  = f2bf(p0);
      pr[16] = f2bf(p1);
      pr[32] = f2bf(p2);
      pr[48] = f2bf(p3);
    }

    Frag pf[2];  // reload P (16x64) as two A-fragments
#pragma unroll
    for (int ks = 0; ks < 2; ++ks) {
      const u16* rp = &Ps[wv][m16 * 64 + ks * 32];
      pf[ks].q[0] = *(const uint4*)(rp + kh * 8);
      pf[ks].q[1] = *(const uint4*)(rp + 16 + kh * 8);
    }
#pragma unroll
    for (int nt = 0; nt < 4; ++nt) {  // O += P * V
#pragma unroll
      for (int ks = 0; ks < 2; ++ks) {
        Frag fv;
        const u16* rp = &Vt[(nt * 16 + m16) * 72 + ks * 32];
        fv.q[0] = *(const uint4*)(rp + kh * 8);
        fv.q[1] = *(const uint4*)(rp + 16 + kh * 8);
        o[nt] = bf16_wmma(pf[ks], fv, o[nt]);
      }
    }
  }

#pragma unroll
  for (int r = 0; r < 8; ++r) {
    float inv = 1.0f / rs[r];
    size_t gm = rowQ + mb + 8 * kh + r;
#pragma unroll
    for (int nt = 0; nt < 4; ++nt)
      O[gm * 512 + head * 64 + nt * 16 + m16] = f2bf(o[nt][r] * inv);
  }
}

// ---------------------------------------------------------------------------
// 5) Final: logmap0(hyp, norm over W) + euc, transpose back to (B,C,H,W)
// ---------------------------------------------------------------------------
__global__ __launch_bounds__(256) void final_kernel(
    const float* __restrict__ Ph, const float* __restrict__ Pe,
    float* __restrict__ out) {
  const int c0 = blockIdx.x * 32, h = blockIdx.y, b = blockIdx.z;
  const int t = threadIdx.x;
  __shared__ float Lh[32][33];
  __shared__ float Le[32][33];
  {
    int c = t & 31, wg = t >> 5;
#pragma unroll
    for (int i = 0; i < 4; ++i) {
      int w = wg * 4 + i;
      size_t idx = ((size_t)b * 1024 + h * 32 + w) * 512 + c0 + c;
      Lh[w][c] = Ph[idx];
      Le[w][c] = Pe[idx];
    }
  }
  __syncthreads();
  {
    int w = t & 31, wg = t >> 5;
#pragma unroll
    for (int i = 0; i < 4; ++i) {
      int c = wg * 4 + i;
      float v = Lh[w][c];
      float ss = v * v;
      ss += __shfl_xor(ss, 1);  ss += __shfl_xor(ss, 2);  ss += __shfl_xor(ss, 4);
      ss += __shfl_xor(ss, 8);  ss += __shfl_xor(ss, 16);
      float n = sqrtf(ss);
      float nc = fminf(n, 1.0f - 1e-5f);
      float sc = atanhf(nc) / fmaxf(n, 1e-15f);
      out[(((size_t)b * 512 + c0 + c) * 32 + h) * 32 + w] = v * sc + Le[w][c];
    }
  }
}

// ---------------------------------------------------------------------------
// Weight conversion: fp32 (K x J) -> bf16 TRANSPOSED (J x K), with optional
// 1/sqrt(hd) pre-scale of the first `scaleRows` output rows (the Q block).
// ---------------------------------------------------------------------------
__global__ __launch_bounds__(256) void cvt_wt_kernel(
    const float* __restrict__ W, u16* __restrict__ Wt, int J, int scaleRows) {
  const int k0 = blockIdx.x * 32, j0 = blockIdx.y * 32;
  const int t = threadIdx.x, c = t & 31, wg = t >> 5;
  __shared__ float T[32][33];
#pragma unroll
  for (int i = 0; i < 4; ++i) {
    int k = wg * 4 + i;
    T[k][c] = W[(size_t)(k0 + k) * J + j0 + c];
  }
  __syncthreads();
#pragma unroll
  for (int i = 0; i < 4; ++i) {
    int jl = wg * 4 + i;
    int j = j0 + jl;
    float v = T[c][jl];
    if (j < scaleRows) v *= 0.125f;
    Wt[(size_t)j * 512 + k0 + c] = f2bf(v);
  }
}

__global__ __launch_bounds__(256) void bias_scale_kernel(
    const float* __restrict__ in, float* __restrict__ out, int n, int nscale) {
  for (int i = blockIdx.x * 256 + threadIdx.x; i < n; i += gridDim.x * 256)
    out[i] = in[i] * ((i < nscale) ? 0.125f : 1.0f);
}

// ---------------------------------------------------------------------------
extern "C" void kernel_launch(void* const* d_in, const int* in_sizes, int n_in,
                              void* d_out, int out_size, void* d_ws, size_t ws_size,
                              hipStream_t stream) {
  (void)in_sizes; (void)n_in; (void)out_size; (void)ws_size;
  const float* x       = (const float*)d_in[0];
  // d_in[1] = current_step (unused)
  const float* w1      = (const float*)d_in[2];
  const float* b1      = (const float*)d_in[3];
  const float* w2      = (const float*)d_in[4];
  const float* b2      = (const float*)d_in[5];
  const float* hw_qkv  = (const float*)d_in[6];
  const float* hb_qkv  = (const float*)d_in[7];
  const float* hw_o    = (const float*)d_in[8];
  const float* hb_o    = (const float*)d_in[9];
  const float* ew_qkv  = (const float*)d_in[10];
  const float* eb_qkv  = (const float*)d_in[11];
  const float* ew_o    = (const float*)d_in[12];
  const float* eb_o    = (const float*)d_in[13];
  float* out = (float*)d_out;

  char* ws = (char*)d_ws;
  size_t off = 0;
  auto take = [&](size_t bytes) -> char* {
    char* p = ws + off;
    off = (off + bytes + 255) & ~(size_t)255;
    return p;
  };
  float* mask = (float*)take((size_t)8 * 512 * 4);
  u16* Wq_h  = (u16*)take((size_t)1536 * 512 * 2);   // transposed (J x K)
  u16* Wo_h  = (u16*)take((size_t)512 * 512 * 2);
  u16* Wq_e  = (u16*)take((size_t)1536 * 512 * 2);
  u16* Wo_e  = (u16*)take((size_t)512 * 512 * 2);
  float* bq_h = (float*)take((size_t)1536 * 4);      // q-scaled qkv bias
  float* bq_e = (float*)take((size_t)1536 * 4);
  u16* Ah   = (u16*)take((size_t)8192 * 512 * 2);
  u16* Ae   = (u16*)take((size_t)8192 * 512 * 2);
  u16* QKV  = (u16*)take((size_t)8192 * 1536 * 2);   // reused per branch
  u16* Ob   = (u16*)take((size_t)8192 * 512 * 2);    // reused per branch
  float* Ph = (float*)take((size_t)8192 * 512 * 4);
  float* Pe = (float*)take((size_t)8192 * 512 * 4);

  // weight conversion + transpose (Q block pre-scaled by 1/sqrt(64))
  cvt_wt_kernel<<<dim3(16, 48), 256, 0, stream>>>(hw_qkv, Wq_h, 1536, 512);
  cvt_wt_kernel<<<dim3(16, 16), 256, 0, stream>>>(hw_o,   Wo_h, 512, 0);
  cvt_wt_kernel<<<dim3(16, 48), 256, 0, stream>>>(ew_qkv, Wq_e, 1536, 512);
  cvt_wt_kernel<<<dim3(16, 16), 256, 0, stream>>>(ew_o,   Wo_e, 512, 0);
  bias_scale_kernel<<<6, 256, 0, stream>>>(hb_qkv, bq_h, 1536, 512);
  bias_scale_kernel<<<6, 256, 0, stream>>>(eb_qkv, bq_e, 1536, 512);

  // gate + split/expmap0/transpose
  gate_kernel<<<8, 256, 0, stream>>>(x, w1, b1, w2, b2, mask);
  prep_kernel<<<dim3(16, 32, 8), 256, 0, stream>>>(x, mask, Ah, Ae);

  // hyperbolic branch
  gemm_bias_kernel<1536, false><<<dim3(12, 64), 256, 0, stream>>>(Ah, Wq_h, bq_h, QKV);
  flash_kernel<<<dim3(8, 8, 8), 256, 0, stream>>>(QKV, Ob);
  gemm_bias_kernel<512, true><<<dim3(4, 64), 256, 0, stream>>>(Ob, Wo_h, hb_o, Ph);

  // euclidean branch
  gemm_bias_kernel<1536, false><<<dim3(12, 64), 256, 0, stream>>>(Ae, Wq_e, eb_qkv, QKV);
  flash_kernel<<<dim3(8, 8, 8), 256, 0, stream>>>(QKV, Ob);
  gemm_bias_kernel<512, true><<<dim3(4, 64), 256, 0, stream>>>(Ob, Wo_e, eb_o, Pe);

  // logmap0 + sum + transpose back
  final_kernel<<<dim3(16, 32, 8), 256, 0, stream>>>(Ph, Pe, out);
}
